// OptimizedGNN_10874857193789
// MI455X (gfx1250) — matrix operand order repaired and checked by reading, not compiled, not run on hardware
//
#include <hip/hip_runtime.h>

#define HEADS   4
#define HID     128
#define HC      512     // HEADS*HID
#define IN_DIM  384
#define OUT_DIM 384

typedef __attribute__((ext_vector_type(16))) __bf16 v16bf;
typedef __attribute__((ext_vector_type(8)))  float  v8f;

union FragB  { v16bf v; uint4 q[2]; };
union Frag16 { uint4 q[2]; unsigned short s[16]; };

__device__ __forceinline__ unsigned short f2bf(float f) {
    unsigned u = __float_as_uint(f);
    unsigned r = (u + 0x7FFFu + ((u >> 16) & 1u)) >> 16;   // round-to-nearest-even
    return (unsigned short)r;
}
__device__ __forceinline__ float bf2f(unsigned short h) {
    return __uint_as_float(((unsigned)h) << 16);
}
// Monotonic float<->uint encoding so unsigned atomicMax == float max
__device__ __forceinline__ unsigned encf(float f) {
    unsigned u = __float_as_uint(f);
    return (u & 0x80000000u) ? ~u : (u | 0x80000000u);
}
__device__ __forceinline__ float decf(unsigned e) {
    unsigned u = (e & 0x80000000u) ? (e & 0x7FFFFFFFu) : ~e;
    return __uint_as_float(u);
}

// ---------------- conversion / setup kernels ----------------

__global__ void k_cvt_f32_bf16(const float* __restrict__ src,
                               unsigned short* __restrict__ dst, int n) {
    int t = blockIdx.x * blockDim.x + threadIdx.x;
    if (t < n) dst[t] = f2bf(src[t]);
}

// W [K,Nc] fp32 row-major  ->  Wt [Nc,K] bf16 row-major
__global__ void k_transW(const float* __restrict__ W,
                         unsigned short* __restrict__ Wt, int K, int Nc) {
    int t = blockIdx.x * blockDim.x + threadIdx.x;
    if (t >= K * Nc) return;
    int k = t / Nc, nc = t - k * Nc;
    Wt[(size_t)nc * K + k] = f2bf(W[t]);
}

__global__ void k_init(float* __restrict__ acc, float* __restrict__ denom,
                       unsigned* __restrict__ nmax, int N) {
    int t = blockIdx.x * blockDim.x + threadIdx.x;
    if (t < N * HC) acc[t] = 0.f;
    if (t < N * HEADS) { denom[t] = 0.f; nmax[t] = 0u; }
}

// ---------------- bf16 WMMA GEMM: C[M,Nc] = A[M,K] x B[K,Nc] ----------------
// A: bf16 row-major [M,K]; Bt: bf16 [Nc,K] (pre-transposed weights).
// Each wave computes a 32(M) x 64(N) strip (8 WMMA accumulators); block = 8
// waves => 256 x 64 tile. B fragments are reused across the two 16-row halves
// (1.5 b128 loads per WMMA). M is assumed a multiple of 16; the high half of
// the last strip is computed on a clamped row and its store is predicated
// (wave-uniform, EXEC stays all-1s through every WMMA).
// If outb != null: write bf16 (no bias). Else: write fp32 + bias (FC layer).
__global__ __launch_bounds__(256) void k_gemm_bf16(
    const unsigned short* __restrict__ A, const unsigned short* __restrict__ Bt,
    int M, int K, int Nc,
    unsigned short* __restrict__ outb, float* __restrict__ outf,
    const float* __restrict__ bias)
{
    const int wave = threadIdx.x >> 5;
    const int lane = threadIdx.x & 31;
    const int row0 = blockIdx.x * 256 + wave * 32;
    if (row0 >= M) return;                       // wave-uniform
    const int col0 = blockIdx.y * 64;
    const int m  = lane & 15;
    const int hf = lane >> 4;

    const bool hiValid = (row0 + 16 < M);
    const int  rowH    = hiValid ? (row0 + 16) : (M - 16);  // clamp duplicate

    v8f cL0 = {}, cL1 = {}, cL2 = {}, cL3 = {};
    v8f cH0 = {}, cH1 = {}, cH2 = {}, cH3 = {};

    const unsigned short* arowL = A + (size_t)(row0 + m) * K;
    const unsigned short* arowH = A + (size_t)(rowH + m) * K;
    const unsigned short* b0r = Bt + (size_t)(col0 +  0 + m) * K + hf * 16;
    const unsigned short* b1r = Bt + (size_t)(col0 + 16 + m) * K + hf * 16;
    const unsigned short* b2r = Bt + (size_t)(col0 + 32 + m) * K + hf * 16;
    const unsigned short* b3r = Bt + (size_t)(col0 + 48 + m) * K + hf * 16;

    for (int k0 = 0; k0 < K; k0 += 32) {
        FragB aL, aH, b;
        // A fragments (16x32 bf16): elems 0..15 -> K=k0+hf*16+e
        const unsigned short* apL = arowL + k0 + hf * 16;
        const unsigned short* apH = arowH + k0 + hf * 16;
        aL.q[0] = *(const uint4*)(apL);
        aL.q[1] = *(const uint4*)(apL + 8);
        aH.q[0] = *(const uint4*)(apH);
        aH.q[1] = *(const uint4*)(apH + 8);
        // B fragments (32x16 bf16): col = m, K = k0 + hf*16 + e (contiguous in Bt)
        b.q[0] = *(const uint4*)(b0r + k0); b.q[1] = *(const uint4*)(b0r + k0 + 8);
        cL0 = __builtin_amdgcn_wmma_f32_16x16x32_bf16(false, aL.v, false, b.v,
                                                      (short)0, cL0, false, false);
        cH0 = __builtin_amdgcn_wmma_f32_16x16x32_bf16(false, aH.v, false, b.v,
                                                      (short)0, cH0, false, false);
        b.q[0] = *(const uint4*)(b1r + k0); b.q[1] = *(const uint4*)(b1r + k0 + 8);
        cL1 = __builtin_amdgcn_wmma_f32_16x16x32_bf16(false, aL.v, false, b.v,
                                                      (short)0, cL1, false, false);
        cH1 = __builtin_amdgcn_wmma_f32_16x16x32_bf16(false, aH.v, false, b.v,
                                                      (short)0, cH1, false, false);
        b.q[0] = *(const uint4*)(b2r + k0); b.q[1] = *(const uint4*)(b2r + k0 + 8);
        cL2 = __builtin_amdgcn_wmma_f32_16x16x32_bf16(false, aL.v, false, b.v,
                                                      (short)0, cL2, false, false);
        cH2 = __builtin_amdgcn_wmma_f32_16x16x32_bf16(false, aH.v, false, b.v,
                                                      (short)0, cH2, false, false);
        b.q[0] = *(const uint4*)(b3r + k0); b.q[1] = *(const uint4*)(b3r + k0 + 8);
        cL3 = __builtin_amdgcn_wmma_f32_16x16x32_bf16(false, aL.v, false, b.v,
                                                      (short)0, cL3, false, false);
        cH3 = __builtin_amdgcn_wmma_f32_16x16x32_bf16(false, aH.v, false, b.v,
                                                      (short)0, cH3, false, false);
    }

    // C/D layout: col = lane&15 (+16*nt), row = strip_row + (lane>>4)*8 + r
    const int ocol = col0 + m;
    const int oL   = row0 + hf * 8;
    const int oH   = rowH + hf * 8;
    if (outb) {
        #pragma unroll
        for (int r = 0; r < 8; ++r) {
            size_t bL = (size_t)(oL + r) * Nc + ocol;
            outb[bL     ] = f2bf(cL0[r]);
            outb[bL + 16] = f2bf(cL1[r]);
            outb[bL + 32] = f2bf(cL2[r]);
            outb[bL + 48] = f2bf(cL3[r]);
        }
        if (hiValid) {
            #pragma unroll
            for (int r = 0; r < 8; ++r) {
                size_t bH = (size_t)(oH + r) * Nc + ocol;
                outb[bH     ] = f2bf(cH0[r]);
                outb[bH + 16] = f2bf(cH1[r]);
                outb[bH + 32] = f2bf(cH2[r]);
                outb[bH + 48] = f2bf(cH3[r]);
            }
        }
    } else {
        const float bia0 = bias[ocol], bia1 = bias[ocol + 16];
        const float bia2 = bias[ocol + 32], bia3 = bias[ocol + 48];
        #pragma unroll
        for (int r = 0; r < 8; ++r) {
            size_t bL = (size_t)(oL + r) * Nc + ocol;
            outf[bL     ] = cL0[r] + bia0;
            outf[bL + 16] = cL1[r] + bia1;
            outf[bL + 32] = cL2[r] + bia2;
            outf[bL + 48] = cL3[r] + bia3;
        }
        if (hiValid) {
            #pragma unroll
            for (int r = 0; r < 8; ++r) {
                size_t bH = (size_t)(oH + r) * Nc + ocol;
                outf[bH     ] = cH0[r] + bia0;
                outf[bH + 16] = cH1[r] + bia1;
                outf[bH + 32] = cH2[r] + bia2;
                outf[bH + 48] = cH3[r] + bia3;
            }
        }
    }
}

// ---------------- attention coefficients: a_src/a_dst [N, HEADS] ----------------
// One wave per node: lane handles 16 channels; head = lane/8; 8-lane reduction.
__global__ __launch_bounds__(256) void k_attn(
    const unsigned short* __restrict__ H,
    const float* __restrict__ att_s, const float* __restrict__ att_d,
    float* __restrict__ a_s, float* __restrict__ a_d, int N)
{
    int node = blockIdx.x * 8 + (threadIdx.x >> 5);
    if (node >= N) return;
    int lane = threadIdx.x & 31;
    int head = lane >> 3;
    int cidx = (lane & 7) * 16;              // channel offset within head

    Frag16 fr;
    const unsigned short* hp = H + (size_t)node * HC + head * HID + cidx;
    fr.q[0] = *(const uint4*)(hp);
    fr.q[1] = *(const uint4*)(hp + 8);

    const float* ws = att_s + head * HID + cidx;
    const float* wd = att_d + head * HID + cidx;
    float ss = 0.f, sd = 0.f;
    #pragma unroll
    for (int i = 0; i < 16; ++i) {
        float f = bf2f(fr.s[i]);
        ss += f * ws[i];
        sd += f * wd[i];
    }
    ss += __shfl_xor(ss, 1); ss += __shfl_xor(ss, 2); ss += __shfl_xor(ss, 4);
    sd += __shfl_xor(sd, 1); sd += __shfl_xor(sd, 2); sd += __shfl_xor(sd, 4);
    if ((lane & 7) == 0) {
        a_s[node * HEADS + head] = ss;
        a_d[node * HEADS + head] = sd;
    }
}

// ---------------- edge pass 1: logits + segment max (encoded atomicMax) ----------------
__global__ void k_edge1(const int* __restrict__ ei, int E, int EP,
                        const float* __restrict__ a_s, const float* __restrict__ a_d,
                        float* __restrict__ ev, unsigned* __restrict__ nmax)
{
    int t = blockIdx.x * blockDim.x + threadIdx.x;
    if (t >= EP) return;
    int src = (t < E) ? ei[t]     : (t - E);   // self loops appended
    int dst = (t < E) ? ei[E + t] : (t - E);
    #pragma unroll
    for (int h = 0; h < HEADS; ++h) {
        float e = a_s[src * HEADS + h] + a_d[dst * HEADS + h];
        e = (e > 0.f) ? e : 0.2f * e;          // leaky relu
        ev[(size_t)t * HEADS + h] = e;
        atomicMax(&nmax[dst * HEADS + h], encf(e));
    }
}

// ---------------- edge pass 2: exp, denom, unnormalized scatter-sum ----------------
// One wave per edge; lane covers 16 of 512 channels (head = lane/8).
__global__ __launch_bounds__(256) void k_edge2(
    const int* __restrict__ ei, int E, int EP,
    const float* __restrict__ ev, const unsigned* __restrict__ nmax,
    float* __restrict__ denom, const unsigned short* __restrict__ H,
    float* __restrict__ acc)
{
    int idx = blockIdx.x * 8 + (threadIdx.x >> 5);
    if (idx >= EP) return;
    int lane = threadIdx.x & 31;
    int src = (idx < E) ? ei[idx]     : (idx - E);
    int dst = (idx < E) ? ei[E + idx] : (idx - E);
    int head = lane >> 3;

    float e = ev[(size_t)idx * HEADS + head];
    float p = __expf(e - decf(nmax[dst * HEADS + head]));
    if ((lane & 7) == 0) atomicAdd(&denom[dst * HEADS + head], p);

    Frag16 fr;
    const unsigned short* hp = H + (size_t)src * HC + lane * 16;
    fr.q[0] = *(const uint4*)(hp);
    fr.q[1] = *(const uint4*)(hp + 8);
    float* ap = acc + (size_t)dst * HC + lane * 16;
    #pragma unroll
    for (int i = 0; i < 16; ++i)
        atomicAdd(ap + i, p * bf2f(fr.s[i]));
}

// ---------------- finalize: normalize, +bias, relu, -> bf16 for next GEMM ----------------
__global__ void k_finalize(const float* __restrict__ acc, const float* __restrict__ denom,
                           const float* __restrict__ bias,
                           unsigned short* __restrict__ Ab, int N)
{
    int t = blockIdx.x * blockDim.x + threadIdx.x;
    if (t >= N * HC) return;
    int n = t >> 9;                 // /512
    int c = t & (HC - 1);
    int h = c >> 7;                 // /128
    float v = acc[t] / denom[n * HEADS + h] + bias[c];
    Ab[t] = f2bf(fmaxf(v, 0.f));
}

// ---------------- host launch ----------------
extern "C" void kernel_launch(void* const* d_in, const int* in_sizes, int n_in,
                              void* d_out, int out_size, void* d_ws, size_t ws_size,
                              hipStream_t stream) {
    const float* x   = (const float*)d_in[0];
    const int*   ei  = (const int*)  d_in[1];
    const float* W1  = (const float*)d_in[2];
    const float* a1s = (const float*)d_in[3];
    const float* a1d = (const float*)d_in[4];
    const float* b1  = (const float*)d_in[5];
    const float* W2  = (const float*)d_in[6];
    const float* a2s = (const float*)d_in[7];
    const float* a2d = (const float*)d_in[8];
    const float* b2  = (const float*)d_in[9];
    const float* Wfc = (const float*)d_in[10];
    const float* bfc = (const float*)d_in[11];

    const int N  = in_sizes[0] / IN_DIM;
    const int E  = in_sizes[1] / 2;
    const int EP = E + N;                  // with self loops

    // workspace carve-up (256B aligned)
    char* p = (char*)d_ws;
    auto alloc = [&](size_t bytes) -> char* {
        char* r = p; p += (bytes + 255) & ~(size_t)255; return r;
    };
    unsigned short* Ab   = (unsigned short*)alloc((size_t)N * HC * 2);  // GEMM input
    unsigned short* Hb   = (unsigned short*)alloc((size_t)N * HC * 2);  // GEMM output h
    float*          acc  = (float*)         alloc((size_t)N * HC * 4);
    unsigned short* W1t  = (unsigned short*)alloc((size_t)HC * IN_DIM * 2);
    unsigned short* W2t  = (unsigned short*)alloc((size_t)HC * HC * 2);
    unsigned short* Wft  = (unsigned short*)alloc((size_t)OUT_DIM * HC * 2);
    float*    a_s   = (float*)   alloc((size_t)N * HEADS * 4);
    float*    a_d   = (float*)   alloc((size_t)N * HEADS * 4);
    float*    denom = (float*)   alloc((size_t)N * HEADS * 4);
    unsigned* nmax  = (unsigned*)alloc((size_t)N * HEADS * 4);
    float*    ev    = (float*)   alloc((size_t)EP * HEADS * 4);

    const int BT = 256;
    // input + weight conversion
    k_cvt_f32_bf16<<<(N * IN_DIM + BT - 1) / BT, BT, 0, stream>>>(x, Ab, N * IN_DIM);
    k_transW<<<(IN_DIM * HC + BT - 1) / BT, BT, 0, stream>>>(W1, W1t, IN_DIM, HC);
    k_transW<<<(HC * HC + BT - 1) / BT, BT, 0, stream>>>(W2, W2t, HC, HC);
    k_transW<<<(HC * OUT_DIM + BT - 1) / BT, BT, 0, stream>>>(Wfc, Wft, HC, OUT_DIM);

    const unsigned short* Wts[2]  = {W1t, W2t};
    const int             Ks[2]   = {IN_DIM, HC};
    const float*          asw[2]  = {a1s, a2s};
    const float*          adw[2]  = {a1d, a2d};
    const float*          bws[2]  = {b1, b2};

    for (int l = 0; l < 2; ++l) {
        k_init<<<(N * HC + BT - 1) / BT, BT, 0, stream>>>(acc, denom, nmax, N);
        dim3 gg((N + 255) / 256, HC / 64);
        k_gemm_bf16<<<gg, BT, 0, stream>>>(Ab, Wts[l], N, Ks[l], HC,
                                           Hb, nullptr, nullptr);
        k_attn<<<(N + 7) / 8, BT, 0, stream>>>(Hb, asw[l], adw[l], a_s, a_d, N);
        k_edge1<<<(EP + BT - 1) / BT, BT, 0, stream>>>(ei, E, EP, a_s, a_d, ev, nmax);
        k_edge2<<<(EP + 7) / 8, BT, 0, stream>>>(ei, E, EP, ev, nmax, denom, Hb, acc);
        k_finalize<<<(N * HC + BT - 1) / BT, BT, 0, stream>>>(acc, denom, bws[l], Ab, N);
    }

    // final FC: d_out[N, OUT_DIM] = (Ab @ Wfc) + bfc, fp32 out
    dim3 gf((N + 255) / 256, OUT_DIM / 64);
    k_gemm_bf16<<<gf, BT, 0, stream>>>(Ab, Wft, N, HC, OUT_DIM,
                                       nullptr, (float*)d_out, bfc);
}
